// SDE_45105746543093
// MI455X (gfx1250) — matrix operand order repaired and checked by reading, not compiled, not run on hardware
//
#include <hip/hip_runtime.h>

typedef __attribute__((ext_vector_type(16))) __bf16 v16bf;
typedef __attribute__((ext_vector_type(8)))  __bf16 v8bf;
typedef __attribute__((ext_vector_type(8)))  float  v8f;

#define NN 64
#define CC 256
#define TT 64
#define VV 25
#define OO 512      // 2*out_planes
#define VP 32       // padded V
#define REL_W 49    // 2*V-1

// ---------------------------------------------------------------------------
// K1: y[n,c,v] = mean_t x[n,c,t,v]; store TRANSPOSED as bf16: ybf[n][v][c]
// (v padded to VP with zeros). Reads of x stay coalesced (v fastest); the
// scattered bf16 writes are only 1 MB total.
// ---------------------------------------------------------------------------
__global__ void k_mean(const float* __restrict__ x, __bf16* __restrict__ ybf) {
    int idx = blockIdx.x * blockDim.x + threadIdx.x;    // over NN*CC*VP
    if (idx >= NN * CC * VP) return;
    int v = idx % VP;
    int c = (idx / VP) % CC;
    int n = idx / (VP * CC);
    float s = 0.f;
    if (v < VV) {
        const float* p = x + ((size_t)(n * CC + c) * TT) * VV + v;
        #pragma unroll 8
        for (int t = 0; t < TT; ++t) s += p[t * VV];
        s *= (1.0f / TT);
    }
    ybf[(size_t)n * VP * CC + (size_t)v * CC + c] = (__bf16)s;
}

// K2: convert W_qkv (512x256, row-major) to bf16
__global__ void k_cvt_w(const float* __restrict__ w, __bf16* __restrict__ wbf) {
    int idx = blockIdx.x * blockDim.x + threadIdx.x;
    if (idx < OO * CC) wbf[idx] = (__bf16)w[idx];
}

// ---------------------------------------------------------------------------
// K3: qkv_raw[n,o,v] = sum_c W[o,c] * y[n,c,v]   via v_wmma_f32_16x16x32_bf16
// One wave per 16x16 output tile; K-loop of 8 WMMA steps (K = 256).
// A frag (16x32 bf16): lane<16 -> row M=lane,    halves = K {0..7,16..23};
//                      lane>=16 -> row M=lane-16, halves = K {8..15,24..31}.
// B frag (32x16 bf16): lane<16 -> col N=lane,    halves = K {0..7,16..23};
//                      lane>=16 -> col N=lane-16, halves = K {8..15,24..31}.
// Both fragments are contiguous in memory (W row-major, y stored transposed),
// so each is exactly two 16-byte loads.
// C/D: VGPR r, lane l -> M = r + (l>=16 ? 8 : 0), N = l&15.
// ---------------------------------------------------------------------------
__global__ void k_qkv_wmma(const __bf16* __restrict__ wbf,
                           const __bf16* __restrict__ ybf,
                           float* __restrict__ qkv) {
    int lane = threadIdx.x & 31;
    int wave = threadIdx.x >> 5;
    int tile = blockIdx.x * 8 + wave;    // 4096 tiles = 64 n * 32 mtiles * 2 ntiles
    int n     = tile >> 6;
    int tt    = tile & 63;
    int mtile = tt >> 1;
    int ntile = tt & 1;

    const __bf16* ybase = ybf + (size_t)n * VP * CC;
    int row  = mtile * 16 + (lane & 15);
    int kofA = (lane >= 16) ? 8 : 0;
    int col  = ntile * 16 + (lane & 15);
    int kofB = (lane >= 16) ? 8 : 0;

    const __bf16* wrow = wbf + (size_t)row * CC;
    const __bf16* ycol = ybase + (size_t)col * CC;

    v8f acc = {};
    #pragma unroll
    for (int k0 = 0; k0 < CC; k0 += 32) {
        v8bf alo = *(const v8bf*)(wrow + k0 + kofA);
        v8bf ahi = *(const v8bf*)(wrow + k0 + kofA + 16);
        v8bf blo = *(const v8bf*)(ycol + k0 + kofB);
        v8bf bhi = *(const v8bf*)(ycol + k0 + kofB + 16);
        v16bf a = __builtin_shufflevector(alo, ahi, 0,1,2,3,4,5,6,7,8,9,10,11,12,13,14,15);
        v16bf b = __builtin_shufflevector(blo, bhi, 0,1,2,3,4,5,6,7,8,9,10,11,12,13,14,15);
        acc = __builtin_amdgcn_wmma_f32_16x16x32_bf16(
            /*neg_a=*/false, a, /*neg_b=*/false, b,
            /*c_mod=*/(short)0, acc, /*reuse_a=*/false, /*reuse_b=*/false);
    }
    int mloc = (lane >= 16) ? 8 : 0;
    float* out = qkv + (size_t)n * OO * VP + (size_t)(mtile * 16 + mloc) * VP + col;
    #pragma unroll
    for (int r = 0; r < 8; ++r) out[r * VP] = acc[r];
}

// ---------------------------------------------------------------------------
// Generic BN-statistics kernel: one block per channel.
// samples at data[ch*cstride + n*nstride + e], n<ncount, e<ecount.
// Writes scale = g/sqrt(var+eps), shift = b - mean*scale.
// ---------------------------------------------------------------------------
__global__ void k_stats(const float* __restrict__ data,
                        const float* __restrict__ g,
                        const float* __restrict__ b,
                        float* __restrict__ scale,
                        float* __restrict__ shift,
                        int nstride, int cstride, int ecount, int ncount) {
    __shared__ float ss[256], sq[256];
    int ch = blockIdx.x;
    const float* base = data + (size_t)ch * cstride;
    float s = 0.f, q = 0.f;
    int tot = ncount * ecount;
    for (int i = threadIdx.x; i < tot; i += blockDim.x) {
        int nn = i / ecount, e = i % ecount;
        float val = base[(size_t)nn * nstride + e];
        s += val; q += val * val;
    }
    ss[threadIdx.x] = s; sq[threadIdx.x] = q;
    __syncthreads();
    for (int off = 128; off > 0; off >>= 1) {
        if ((int)threadIdx.x < off) {
            ss[threadIdx.x] += ss[threadIdx.x + off];
            sq[threadIdx.x] += sq[threadIdx.x + off];
        }
        __syncthreads();
    }
    if (threadIdx.x == 0) {
        float inv = 1.0f / tot;
        float m   = ss[0] * inv;
        float var = sq[0] * inv - m * m;
        float sc  = g[ch] * rsqrtf(var + 1e-5f);
        scale[ch] = sc;
        shift[ch] = b[ch] - m * sc;
    }
}

// ---------------------------------------------------------------------------
// K5: per (n,g): qk / qr / kr scores (25x25 each) -> stacked (N,24,25,25)
//   qk[i,j] = sum_c q[c,i]k[c,j]
//   qr[i,j] = sum_c q[c,i]*rel_q[c, i-j+24]
//   kr[i,j] = sum_c k[c,j]*rel_k[c, j-i+24]   (post-swapaxes form)
// ---------------------------------------------------------------------------
__global__ void k_attn_scores(const float* __restrict__ qkv,
                              const float* __restrict__ scale1,
                              const float* __restrict__ shift1,
                              const float* __restrict__ rel,
                              float* __restrict__ stacked) {
    __shared__ float qs[16][VV], ks[16][VV], rl[32][REL_W];
    int n  = blockIdx.x >> 3;
    int gI = blockIdx.x & 7;
    for (int i = threadIdx.x; i < 32 * VV; i += blockDim.x) {
        int c = i / VV, v = i % VV;
        int o = gI * 64 + c;
        float val = qkv[(size_t)n * OO * VP + (size_t)o * VP + v] * scale1[o] + shift1[o];
        if (c < 16) qs[c][v] = val; else ks[c - 16][v] = val;
    }
    for (int i = threadIdx.x; i < 32 * REL_W; i += blockDim.x)
        rl[i / REL_W][i % REL_W] = rel[i];          // rows 0..15 = q_e, 16..31 = k_e
    __syncthreads();
    float* ob = stacked + (size_t)n * 24 * 625;
    for (int e = threadIdx.x; e < 625; e += blockDim.x) {
        int i = e / VV, j = e % VV;
        int dij = i - j + 24, dji = j - i + 24;
        float qk = 0.f, qr = 0.f, kr = 0.f;
        #pragma unroll
        for (int c = 0; c < 16; ++c) {
            qk += qs[c][i] * ks[c][j];
            qr += qs[c][i] * rl[c][dij];
            kr += ks[c][j] * rl[16 + c][dji];
        }
        ob[(size_t)gI * 625 + e]        = qk;
        ob[(size_t)(8 + gI) * 625 + e]  = qr;
        ob[(size_t)(16 + gI) * 625 + e] = kr;
    }
}

// ---------------------------------------------------------------------------
// K7: per (n,g): BN+sum 3 score channels, softmax over j, then
//   sv[c,i]  = sum_j sim[i,j]*v[c,j]
//   sve[c,i] = sum_j sim[i,j]*rel_v[c, i-j+24]
// scatter into so (N,512,25) with o = (g*32+c)*2 + h
// ---------------------------------------------------------------------------
__global__ void k_attn_out(const float* __restrict__ qkv,
                           const float* __restrict__ scale1,
                           const float* __restrict__ shift1,
                           const float* __restrict__ scale2,
                           const float* __restrict__ shift2,
                           const float* __restrict__ rel,
                           const float* __restrict__ stacked,
                           float* __restrict__ so) {
    __shared__ float sm[VV][VV];
    __shared__ float vs[32][VV];
    __shared__ float rv[32][REL_W];
    int n  = blockIdx.x >> 3;
    int gI = blockIdx.x & 7;
    const float* stb = stacked + (size_t)n * 24 * 625;
    int c0 = gI, c1 = 8 + gI, c2 = 16 + gI;
    float s0 = scale2[c0], h0 = shift2[c0];
    float s1 = scale2[c1], h1 = shift2[c1];
    float s2 = scale2[c2], h2 = shift2[c2];
    for (int e = threadIdx.x; e < 625; e += blockDim.x) {
        float val = stb[c0 * 625 + e] * s0 + h0
                  + stb[c1 * 625 + e] * s1 + h1
                  + stb[c2 * 625 + e] * s2 + h2;
        sm[e / VV][e % VV] = val;
    }
    for (int i = threadIdx.x; i < 32 * VV; i += blockDim.x) {
        int c = i / VV, j = i % VV;
        int o = gI * 64 + 32 + c;
        vs[c][j] = qkv[(size_t)n * OO * VP + (size_t)o * VP + j] * scale1[o] + shift1[o];
    }
    for (int i = threadIdx.x; i < 32 * REL_W; i += blockDim.x)
        rv[i / REL_W][i % REL_W] = rel[32 * REL_W + i];   // rows 32..63 = v_e
    __syncthreads();
    if ((int)threadIdx.x < VV) {                          // softmax per row
        int i = threadIdx.x;
        float mx = -1e30f;
        for (int j = 0; j < VV; ++j) mx = fmaxf(mx, sm[i][j]);
        float sum = 0.f;
        for (int j = 0; j < VV; ++j) { float ex = __expf(sm[i][j] - mx); sm[i][j] = ex; sum += ex; }
        float inv = 1.0f / sum;
        for (int j = 0; j < VV; ++j) sm[i][j] *= inv;
    }
    __syncthreads();
    float* ob = so + (size_t)n * OO * VV;
    for (int e = threadIdx.x; e < 1600; e += blockDim.x) {
        int h = e / 800;
        int rem = e % 800;
        int c = rem / VV, i = rem % VV;
        float acc = 0.f;
        if (h == 0) {
            #pragma unroll
            for (int j = 0; j < VV; ++j) acc += sm[i][j] * vs[c][j];
        } else {
            #pragma unroll
            for (int j = 0; j < VV; ++j) acc += sm[i][j] * rv[c][i - j + 24];
        }
        int o = (gI * 32 + c) * 2 + h;
        ob[(size_t)o * VV + i] = acc;
    }
}

// K9: gate[n,c,v] = 1 + sigmoid( bn3(so[2c]) + bn3(so[2c+1]) )
__global__ void k_gate(const float* __restrict__ so,
                       const float* __restrict__ scale3,
                       const float* __restrict__ shift3,
                       float* __restrict__ gate) {
    int idx = blockIdx.x * blockDim.x + threadIdx.x;
    if (idx >= NN * CC * VV) return;
    int v = idx % VV;
    int c = (idx / VV) % CC;
    int n = idx / (VV * CC);
    const float* base = so + (size_t)n * OO * VV;
    int o0 = 2 * c, o1 = 2 * c + 1;
    float val = base[(size_t)o0 * VV + v] * scale3[o0] + shift3[o0]
              + base[(size_t)o1 * VV + v] * scale3[o1] + shift3[o1];
    gate[idx] = 1.0f + 1.0f / (1.0f + __expf(-val));
}

// ---------------------------------------------------------------------------
// K10: out = x * gate (gate broadcast over T). float4-vectorized streaming
// pass: total elements divisible by 4, and each (n,c) block of T*V = 1600
// elements is divisible by 4, so a float4 never crosses an (n,c) boundary.
// ---------------------------------------------------------------------------
__global__ void k_final(const float4* __restrict__ x4,
                        const float* __restrict__ gate,
                        float4* __restrict__ out4) {
    size_t idx = (size_t)blockIdx.x * blockDim.x + threadIdx.x;
    const size_t total4 = (size_t)NN * CC * TT * VV / 4;
    if (idx >= total4) return;
    size_t e  = idx * 4;
    size_t nc = e / ((size_t)TT * VV);        // n*CC + c
    int    r  = (int)(e % ((size_t)TT * VV)); // offset within (n,c) block
    const float* gp = gate + nc * VV;
    float4 xv = x4[idx];
    float4 ov;
    ov.x = xv.x * gp[(r + 0) % VV];
    ov.y = xv.y * gp[(r + 1) % VV];
    ov.z = xv.z * gp[(r + 2) % VV];
    ov.w = xv.w * gp[(r + 3) % VV];
    out4[idx] = ov;
}

extern "C" void kernel_launch(void* const* d_in, const int* in_sizes, int n_in,
                              void* d_out, int out_size, void* d_ws, size_t ws_size,
                              hipStream_t stream) {
    (void)in_sizes; (void)n_in; (void)out_size; (void)ws_size;
    const float* x     = (const float*)d_in[0];
    const float* w_qkv = (const float*)d_in[1];
    const float* rel   = (const float*)d_in[2];
    const float* g1 = (const float*)d_in[3];
    const float* b1 = (const float*)d_in[4];
    const float* g2 = (const float*)d_in[5];
    const float* b2 = (const float*)d_in[6];
    const float* g3 = (const float*)d_in[7];
    const float* b3 = (const float*)d_in[8];
    float* out = (float*)d_out;

    char* ws = (char*)d_ws;
    size_t off = 0;
    auto alloc = [&](size_t bytes) -> void* {
        void* p = ws + off;
        off += (bytes + 255) & ~(size_t)255;
        return p;
    };
    __bf16* ybf     = (__bf16*)alloc((size_t)NN * VP * CC * 2);     // 1 MB (transposed)
    __bf16* wbf     = (__bf16*)alloc((size_t)OO * CC * 2);          // 256 KB
    float*  qkv     = (float*) alloc((size_t)NN * OO * VP * 4);     // 4 MB
    float*  scale1  = (float*) alloc(OO * 4);
    float*  shift1  = (float*) alloc(OO * 4);
    float*  stacked = (float*) alloc((size_t)NN * 24 * 625 * 4);    // 3.84 MB
    float*  scale2  = (float*) alloc(24 * 4);
    float*  shift2  = (float*) alloc(24 * 4);
    float*  so      = (float*) alloc((size_t)NN * OO * VV * 4);     // 3.28 MB
    float*  scale3  = (float*) alloc(OO * 4);
    float*  shift3  = (float*) alloc(OO * 4);
    float*  gate    = (float*) alloc((size_t)NN * CC * VV * 4);     // 1.6 MB

    // 1. temporal mean (streams x once) -> transposed bf16 y; convert W
    k_mean<<<(NN * CC * VP + 255) / 256, 256, 0, stream>>>(x, ybf);
    k_cvt_w<<<(OO * CC + 255) / 256, 256, 0, stream>>>(w_qkv, wbf);
    // 2. qkv GEMM via WMMA bf16 (4096 tiles, 8 waves/block)
    k_qkv_wmma<<<512, 256, 0, stream>>>(wbf, ybf, qkv);
    // 3. BN(qkv) stats over (n,v) per o-channel
    k_stats<<<OO, 256, 0, stream>>>(qkv, g1, b1, scale1, shift1,
                                    OO * VP, VP, VV, NN);
    // 4. qk/qr/kr scores
    k_attn_scores<<<NN * 8, 256, 0, stream>>>(qkv, scale1, shift1, rel, stacked);
    // 5. BN(sim) stats over (n,i,j) per 24 channels
    k_stats<<<24, 256, 0, stream>>>(stacked, g2, b2, scale2, shift2,
                                    24 * 625, 625, 625, NN);
    // 6. BN+sum+softmax+sv/sve
    k_attn_out<<<NN * 8, 256, 0, stream>>>(qkv, scale1, shift1,
                                           scale2, shift2, rel, stacked, so);
    // 7. BN(out) stats over (n,v) per o-channel
    k_stats<<<OO, 256, 0, stream>>>(so, g3, b3, scale3, shift3,
                                    OO * VV, VV, VV, NN);
    // 8. gate = 1 + sigmoid(...)
    k_gate<<<(NN * CC * VV + 255) / 256, 256, 0, stream>>>(so, scale3, shift3, gate);
    // 9. out = x * gate (streams x + out), float4-vectorized
    const size_t total4 = (size_t)NN * CC * TT * VV / 4;
    k_final<<<(unsigned)((total4 + 255) / 256), 256, 0, stream>>>(
        (const float4*)x, gate, (float4*)out);
}
